// DenseAttentionSelector_22754736734717
// MI455X (gfx1250) — matrix operand order repaired and checked by reading, not compiled, not run on hardware
//
#include <hip/hip_runtime.h>
#include <hip/hip_bf16.h>

typedef __attribute__((ext_vector_type(16))) __bf16 v16bf;
typedef __attribute__((ext_vector_type(8)))  __bf16 v8bf;
typedef __attribute__((ext_vector_type(8)))  float  v8f;

#define WMMA_BF16(a, b, c) \
  __builtin_amdgcn_wmma_f32_16x16x32_bf16(false, (a), false, (b), (short)0, (c), false, false)

// ---------------- CDNA5 async global->LDS copy (ASYNCcnt-tracked) ----------------
// VDST = VGPR holding wave-relative LDS byte address; VADDR = 64-bit global address.
__device__ __forceinline__ void async_copy_b128(unsigned lds_addr, const void* gptr) {
  asm volatile("global_load_async_to_lds_b128 %0, %1, off"
               :: "v"(lds_addr), "v"((unsigned long long)gptr)
               : "memory");
}

#if defined(__has_builtin)
#if __has_builtin(__builtin_amdgcn_s_wait_asynccnt)
#define WAIT_ASYNC0() __builtin_amdgcn_s_wait_asynccnt(0)
#endif
#endif
#ifndef WAIT_ASYNC0
#define WAIT_ASYNC0() asm volatile("s_wait_asynccnt 0" ::: "memory")
#endif

// ---------------- operand loaders (ISA 7.12.2 layouts, wave32) ----------------
// A-matrix 16x32 bf16: lane holds row M=lane%16; half=lane/16.
//   elements 0..7  -> K = k0 + half*8 + j        (VGPR0-3)
//   elements 8..15 -> K = k0 + 16 + half*8 + j   (VGPR4-7)
__device__ __forceinline__ v16bf load_a_f32(const float* __restrict__ src, int ld,
                                            int row0, int k0, int lane) {
  const int half = lane >> 4;
  const float* p = src + (row0 + (lane & 15)) * ld + k0 + half * 8;
  v16bf r;
#pragma unroll
  for (int j = 0; j < 8; ++j) r[j] = (__bf16)p[j];
#pragma unroll
  for (int j = 0; j < 8; ++j) r[j + 8] = (__bf16)p[j + 16];
  return r;
}

__device__ __forceinline__ v16bf load_a_bf16(const __bf16* __restrict__ src, int ld,
                                             int row0, int k0, int lane) {
  const int half = lane >> 4;
  const __bf16* p = src + (row0 + (lane & 15)) * ld + k0 + half * 8;
  v8bf lo = *(const v8bf*)p;
  v8bf hi = *(const v8bf*)(p + 16);
  v16bf r;
#pragma unroll
  for (int j = 0; j < 8; ++j) { r[j] = lo[j]; r[j + 8] = hi[j]; }
  return r;
}

// B-matrix 32x16 bf16 (KxN): lane holds column N=lane%16; half=lane/16.
//   elements 0..7 -> K = k0 + half*8 + j; elements 8..15 -> K = k0 + 16 + half*8 + j
__device__ __forceinline__ v16bf load_b_bf16(const __bf16* __restrict__ src, int ld,
                                             int n0, int k0, int lane) {
  const int half = lane >> 4;
  const __bf16* p = src + (n0 + (lane & 15)) * ld + k0 + half * 8;
  v8bf lo = *(const v8bf*)p;
  v8bf hi = *(const v8bf*)(p + 16);
  v16bf r;
#pragma unroll
  for (int j = 0; j < 8; ++j) { r[j] = lo[j]; r[j + 8] = hi[j]; }
  return r;
}

// ---------------- kernel 0: zero s2 accumulator (2*2048 floats) ----------------
__global__ void dasel_zero_kernel(float* __restrict__ s2) {
  s2[blockIdx.x * 256 + threadIdx.x] = 0.0f;
}

// ---------------- kernel 1: QK projection GEMM (bf16 WMMA + async LDS staging) --
// qk[m][0:1024]    = (x @ Wq^T + bq) * 0.125   (scale 1/sqrt(64) folded in)
// qk[m][1024:2048] = x @ Wk^T + bk
// Block = 8 waves sharing one 64-col group (same n0): W tile (64 rows x 32 K f32,
// 8 KB) is DMA'd into double-buffered LDS via global_load_async_to_lds_b128 and
// consumed by all 8 waves; A streams per-wave from global.
__global__ __launch_bounds__(256) void dasel_proj_kernel(
    const float* __restrict__ x, const float* __restrict__ w,
    const float* __restrict__ bias, __bf16* __restrict__ qk) {
  __shared__ __attribute__((aligned(64))) float wbuf[2][64 * 32];  // 16 KB
  const int lane = threadIdx.x & 31;
  const int wave = threadIdx.x >> 5;
  const int gw = blockIdx.x * 8 + wave;        // 0..8191
  const int m0 = (gw & 255) << 4;              // 256 row tiles of 16
  const int n0 = (gw >> 8) << 6;               // 32 col groups of 64 (uniform per block)
  const int tid = threadIdx.x;

  // stage 8 KB W tile (rows n0..n0+63, cols k0..k0+31) into wbuf[buf]
  auto stage = [&](int buf, int k0) {
#pragma unroll
    for (int c = 0; c < 2; ++c) {
      const int idx4 = tid + c * 256;          // 512 x b128 chunks
      const int r = idx4 >> 3;                 // row in tile (8 chunks per row)
      const int cc = (idx4 & 7) << 2;          // col (floats)
      const float* src = w + (n0 + r) * 1024 + k0 + cc;
      const unsigned dst = (unsigned)(unsigned long long)&wbuf[buf][idx4 * 4];
      async_copy_b128(dst, src);
    }
  };

  stage(0, 0);
  int cur = 0;
  v8f c[4] = {};
  for (int k0 = 0; k0 < 1024; k0 += 32) {
    WAIT_ASYNC0();
    __syncthreads();                            // wbuf[cur] ready for everyone
    if (k0 + 32 < 1024) stage(cur ^ 1, k0 + 32);

    v16bf a = load_a_f32(x, 1024, m0, k0, lane);
#pragma unroll
    for (int t = 0; t < 4; ++t) {
      // B operand from LDS: lane = column N, K = half*8 + j and 16 + half*8 + j
      const float* p = &wbuf[cur][(t * 16 + (lane & 15)) * 32 + (lane >> 4) * 8];
      v16bf bt;
#pragma unroll
      for (int j = 0; j < 16; ++j) bt[j] = (__bf16)p[j + (j & 8)];
      c[t] = WMMA_BF16(a, bt, c[t]);
    }
    __syncthreads();                            // safe to restage this buffer later
    cur ^= 1;
  }

  const int lr = lane & 15;
  const int half = lane >> 4;
#pragma unroll
  for (int t = 0; t < 4; ++t) {
    const int n = n0 + t * 16 + lr;
    const float bs = bias[n];
    const float sc = (n < 1024) ? 0.125f : 1.0f;
#pragma unroll
    for (int r = 0; r < 8; ++r) {
      const int m = m0 + half * 8 + r;          // C layout: M = r + half*8, N = lane%16
      qk[m * 2048 + n] = (__bf16)((c[t][r] + bs) * sc);
    }
  }
}

// ---------------- kernel 2: scores + softmax + per-key mass ----------------
// One WG per (b, h, 16-query tile). 8 waves, each covers 256 keys.
// Dynamic LDS: 16 x 2048 f32 scores (128 KB) -> 2 blocks resident per WGP.
__global__ __launch_bounds__(256) void dasel_attn_kernel(
    const __bf16* __restrict__ qk, float* __restrict__ s2) {
  extern __shared__ float sc[];                 // 16*2048 floats
  __shared__ float invZ[16];
  const int lane = threadIdx.x & 31;
  const int wave = threadIdx.x >> 5;
  const int id = blockIdx.x;
  const int qt = id & 127;
  const int h  = (id >> 7) & 15;
  const int b  = id >> 11;
  const int q0 = qt << 4;
  const int rowQ = b * 2048 + q0;
  const int colQ = h * 64;
  const int colK = 1024 + h * 64;

  // scale already folded into Q; hd=64 -> two K=32 WMMA steps
  v16bf a0 = load_a_bf16(qk, 2048, rowQ, colQ, lane);
  v16bf a1 = load_a_bf16(qk, 2048, rowQ, colQ + 32, lane);

  for (int t = 0; t < 16; ++t) {
    const int key0 = wave * 256 + t * 16;
    v16bf b0 = load_b_bf16(qk, 2048, b * 2048 + key0, colK, lane);
    v16bf b1 = load_b_bf16(qk, 2048, b * 2048 + key0, colK + 32, lane);
    v8f c = {};
    c = WMMA_BF16(a0, b0, c);
    c = WMMA_BF16(a1, b1, c);
    const int nn = key0 + (lane & 15);
    const int half = lane >> 4;
#pragma unroll
    for (int r = 0; r < 8; ++r) sc[(half * 8 + r) * 2048 + nn] = c[r];
  }
  __syncthreads();

  // row softmax (max, exp-in-place, sum); wave handles 2 rows
#pragma unroll
  for (int rr = 0; rr < 2; ++rr) {
    const int q = wave * 2 + rr;
    float* row = sc + q * 2048;
    float m = -1e30f;
    for (int i = lane; i < 2048; i += 32) m = fmaxf(m, row[i]);
#pragma unroll
    for (int off = 16; off; off >>= 1) m = fmaxf(m, __shfl_xor(m, off, 32));
    float z = 0.0f;
    for (int i = lane; i < 2048; i += 32) {
      float e = __expf(row[i] - m);
      row[i] = e;
      z += e;
    }
#pragma unroll
    for (int off = 16; off; off >>= 1) z += __shfl_xor(z, off, 32);
    if (lane == 0) invZ[q] = 1.0f / z;
  }
  __syncthreads();

  // column sums over 16 queries, accumulate into global s2[b][key]
  const int key = threadIdx.x * 8;
  float acc[8] = {};
#pragma unroll
  for (int q = 0; q < 16; ++q) {
    const float iz = invZ[q];
#pragma unroll
    for (int j = 0; j < 8; ++j) acc[j] += sc[q * 2048 + key + j] * iz;
  }
#pragma unroll
  for (int j = 0; j < 8; ++j) atomicAdd(&s2[b * 2048 + key + j], acc[j]);
}

// ---------------- kernel 3: per-batch descending argsort (bitonic, LDS) --------
__global__ __launch_bounds__(1024) void dasel_sort_kernel(
    const float* __restrict__ s2, int* __restrict__ order, float* __restrict__ top_out) {
  __shared__ float sv[2048];
  __shared__ int   si[2048];
  const int b = blockIdx.x;
  const int t = threadIdx.x;
  sv[t] = s2[b * 2048 + t];               si[t] = t;
  sv[t + 1024] = s2[b * 2048 + t + 1024]; si[t + 1024] = t + 1024;
  __syncthreads();

  for (int ksz = 2; ksz <= 2048; ksz <<= 1) {
    for (int j = ksz >> 1; j > 0; j >>= 1) {
      const int i = ((t & ~(j - 1)) << 1) | (t & (j - 1));
      const int p = i + j;
      const bool d0 = (i & ksz) == 0;
      const float vi = sv[i], vp = sv[p];
      const int   ii = si[i], ip = si[p];
      // desired final order: descending value, ascending index on ties
      const bool after = (vi < vp) || (vi == vp && ii > ip);
      if (after == d0) { sv[i] = vp; sv[p] = vi; si[i] = ip; si[p] = ii; }
      __syncthreads();
    }
  }
  order[b * 2048 + t] = si[t];
  order[b * 2048 + t + 1024] = si[t + 1024];
  top_out[b * 1024 + t] = (float)si[t];   // top-k indices into output tail
}

// ---------------- kernel 4: gather x rows into k_set / q_set ----------------
__global__ __launch_bounds__(256) void dasel_gather_kernel(
    const float* __restrict__ x, const int* __restrict__ order,
    float* __restrict__ kset, float* __restrict__ qset) {
  const int id = blockIdx.x;                // b*2048 + j
  const int b = id >> 11;
  const int j = id & 2047;
  const int idx = order[id];
  const float4* src = (const float4*)(x + (size_t)(b * 2048 + idx) * 1024);
  float* dstBase = (j < 1024)
      ? (kset + (size_t)(b * 1024 + j) * 1024)
      : (qset + (size_t)(b * 1024 + (j - 1024)) * 1024);
  ((float4*)dstBase)[threadIdx.x] = src[threadIdx.x];
}

// ---------------- host launcher ----------------
extern "C" void kernel_launch(void* const* d_in, const int* in_sizes, int n_in,
                              void* d_out, int out_size, void* d_ws, size_t ws_size,
                              hipStream_t stream) {
  const float* x    = (const float*)d_in[0];   // (2, 2048, 1024) f32
  const float* w    = (const float*)d_in[1];   // (3072, 1024) f32
  const float* bias = (const float*)d_in[2];   // (3072,) f32
  (void)in_sizes; (void)n_in; (void)out_size; (void)ws_size;

  char* ws = (char*)d_ws;
  __bf16* qk  = (__bf16*)ws;                                       // 4096*2048 bf16 = 16 MB
  float*  s2  = (float*)(ws + (size_t)4096 * 2048 * 2);            // 4096 f32
  int*    ord = (int*)(ws + (size_t)4096 * 2048 * 2 + 4096 * 4);   // 4096 i32

  float* out  = (float*)d_out;
  float* kset = out;                                // 2*1024*1024
  float* qset = out + (size_t)2 * 1024 * 1024;      // 2*1024*1024
  float* top  = out + (size_t)4 * 1024 * 1024;      // 2*1024

  dasel_zero_kernel<<<16, 256, 0, stream>>>(s2);
  dasel_proj_kernel<<<1024, 256, 0, stream>>>(x, w, bias, qk);
  dasel_attn_kernel<<<4096, 256, 16 * 2048 * sizeof(float), stream>>>(qk, s2);
  dasel_sort_kernel<<<2, 1024, 0, stream>>>(s2, ord, top);
  dasel_gather_kernel<<<4096, 256, 0, stream>>>(x, ord, kset, qset);
}
